// MultiHeadAttention4CLKT_21320217657943
// MI455X (gfx1250) — matrix operand order repaired and checked by reading, not compiled, not run on hardware
//
#include <hip/hip_runtime.h>
#include <hip/hip_bf16.h>

// Problem constants from the reference
#define B_  4
#define S_  1024
#define D_  512
#define H_  8
#define DH_ 64

typedef __bf16 bf16;
typedef __attribute__((ext_vector_type(16))) __bf16 v16bf;
typedef __attribute__((ext_vector_type(8)))  __bf16 v8bf;
typedef __attribute__((ext_vector_type(8)))  float  v8f;

static __device__ __forceinline__ v8f wmma_bf16(v16bf a, v16bf b, v8f c) {
  // D = A(16x32 bf16) x B(32x16 bf16) + C(16x16 f32)
  return __builtin_amdgcn_wmma_f32_16x16x32_bf16(false, a, false, b, (short)0, c,
                                                 false, false);
}

// ---- WMMA fragment loaders (bf16 memory, 128-bit vector loads) --------------
// A 16x32: lane L -> row m = L&15, half = L>>4.
// Elements 0..7  hold K = k0 + half*8 + 0..7   (contiguous)
// Elements 8..15 hold K = k0 + 16 + half*8 + 0..7 (contiguous)
static __device__ __forceinline__ v16bf load_a_v(const bf16* __restrict__ src,
                                                 int ld, int row0, int k0, int lane) {
  int m = lane & 15, half = lane >> 4;
  const bf16* p = src + (size_t)(row0 + m) * ld + k0 + half * 8;
  v8bf lo = *(const v8bf*)p;        // 16B aligned
  v8bf hi = *(const v8bf*)(p + 16); // +32 bytes
  v16bf a;
#pragma unroll
  for (int i = 0; i < 8; ++i) { a[i] = lo[i]; a[i + 8] = hi[i]; }
  return a;
}

// B 32x16 where b(k,n) = W[n][k] (W row-major): lane -> col n = L&15,
// half = L>>4; elements 0..7 hold K = half*8 + 0..7, 8..15 hold K = 16 + half*8 + 0..7.
static __device__ __forceinline__ v16bf load_bT_v(const bf16* __restrict__ W,
                                                  int ld, int k0, int n0, int lane) {
  int n = lane & 15, kb = (lane >> 4) * 8;
  const bf16* p = W + (size_t)(n0 + n) * ld + k0 + kb;
  v8bf lo = *(const v8bf*)p;
  v8bf hi = *(const v8bf*)(p + 16);
  v16bf b;
#pragma unroll
  for (int i = 0; i < 8; ++i) { b[i] = lo[i]; b[i + 8] = hi[i]; }
  return b;
}

// A 16x32 from fp32 LDS strip (attn), vectorized float4 loads + pack to bf16
static __device__ __forceinline__ v16bf load_a_lds(const float* __restrict__ sc,
                                                   int ld, int k0, int lane) {
  int m = lane & 15, half = lane >> 4;
  const float* p = sc + (size_t)m * ld + k0 + half * 8;
  float4 x0 = *(const float4*)(p + 0);
  float4 x1 = *(const float4*)(p + 4);
  float4 x2 = *(const float4*)(p + 16);
  float4 x3 = *(const float4*)(p + 20);
  v16bf a;
  a[0]=(__bf16)x0.x; a[1]=(__bf16)x0.y; a[2]=(__bf16)x0.z; a[3]=(__bf16)x0.w;
  a[4]=(__bf16)x1.x; a[5]=(__bf16)x1.y; a[6]=(__bf16)x1.z; a[7]=(__bf16)x1.w;
  a[8]=(__bf16)x2.x; a[9]=(__bf16)x2.y; a[10]=(__bf16)x2.z; a[11]=(__bf16)x2.w;
  a[12]=(__bf16)x3.x; a[13]=(__bf16)x3.y; a[14]=(__bf16)x3.z; a[15]=(__bf16)x3.w;
  return a;
}

// ---- Kernel 0: streaming fp32 -> bf16 conversion (8 elems / thread) ---------
__global__ void f32_to_bf16_kernel(const float* __restrict__ in,
                                   bf16* __restrict__ out, int n8) {
  int i = blockIdx.x * blockDim.x + threadIdx.x;
  if (i >= n8) return;
  const float4* p = (const float4*)in + (size_t)i * 2;
  float4 x0 = p[0], x1 = p[1];
  v8bf o;
  o[0]=(__bf16)x0.x; o[1]=(__bf16)x0.y; o[2]=(__bf16)x0.z; o[3]=(__bf16)x0.w;
  o[4]=(__bf16)x1.x; o[5]=(__bf16)x1.y; o[6]=(__bf16)x1.z; o[7]=(__bf16)x1.w;
  ((v8bf*)out)[i] = o;
}

// ---- Kernel 1: projection GEMM  Y = X @ W^T + bias --------------------------
// One wave per 16x64 output tile (4 WMMA accumulators, A reused 4x).
// mode 0: bf16 head layout [B,H,S,DH] (Q,K)
// mode 2: bf16 transposed head layout [B,H,DH,S] (V)
// mode 1: fp32 row-major (final output)
__global__ void proj_gemm_kernel(const bf16* __restrict__ X,
                                 const bf16* __restrict__ W,
                                 const float* __restrict__ bias,
                                 bf16* __restrict__ outB,
                                 float* __restrict__ outRow,
                                 int Kdim, int Ndim, int mode) {
  int lane = threadIdx.x & 31;
  int r0 = blockIdx.x * 16;
  int n0 = blockIdx.y * 64;
  v8f acc0 = {}, acc1 = {}, acc2 = {}, acc3 = {};
  for (int k0 = 0; k0 < Kdim; k0 += 32) {
    v16bf a = load_a_v(X, Kdim, r0, k0, lane);
    acc0 = wmma_bf16(a, load_bT_v(W, Kdim, k0, n0 +  0, lane), acc0);
    acc1 = wmma_bf16(a, load_bT_v(W, Kdim, k0, n0 + 16, lane), acc1);
    acc2 = wmma_bf16(a, load_bT_v(W, Kdim, k0, n0 + 32, lane), acc2);
    acc3 = wmma_bf16(a, load_bT_v(W, Kdim, k0, n0 + 48, lane), acc3);
  }
  int n = lane & 15, mb = (lane >> 4) * 8;
  v8f accs[4] = {acc0, acc1, acc2, acc3};
#pragma unroll
  for (int t = 0; t < 4; ++t) {
    int col = n0 + 16 * t + n;
    float bv = bias[col];
    int h = col >> 6, dh = col & 63;  // DH = 64
#pragma unroll
    for (int rr = 0; rr < 8; ++rr) {
      int row = r0 + mb + rr;
      int bb = row >> 10, s = row & 1023;  // S = 1024
      float val = accs[t][rr] + bv;
      if (mode == 0) {
        outB[(((size_t)bb * H_ + h) * S_ + s) * DH_ + dh] = (__bf16)val;
      } else if (mode == 2) {
        outB[(((size_t)bb * H_ + h) * DH_ + dh) * S_ + s] = (__bf16)val;
      } else {
        outRow[(size_t)row * Ndim + col] = val;
      }
    }
  }
}

// ---- Kernel 2: fused attention per (b, h, 16-query tile) --------------------
// 256 threads = 8 waves. LDS: 16 x 1024 fp32 score strip + 256 fp32 reductions.
__global__ void attn_kernel(const bf16* __restrict__ qh,
                            const bf16* __restrict__ kh,
                            const bf16* __restrict__ vhT,
                            const float* __restrict__ gammas,
                            float* __restrict__ attn_out,
                            bf16* __restrict__ concat) {
  extern __shared__ float smem[];
  float* sc  = smem;             // [16][S_]
  float* red = smem + 16 * S_;   // [16][16]

  const int qt = blockIdx.x, h = blockIdx.y, b = blockIdx.z;
  const int tid = threadIdx.x, wave = tid >> 5, lane = tid & 31;
  const int q0 = qt * 16;

  const bf16* qbh  = qh  + ((size_t)b * H_ + h) * S_ * DH_;
  const bf16* kbh  = kh  + ((size_t)b * H_ + h) * S_ * DH_;
  const bf16* vbhT = vhT + ((size_t)b * H_ + h) * DH_ * S_;

  // gamma = -softplus(gammas[h])  (numerically stable)
  float g0 = gammas[h];
  float sp = (g0 > 0.f) ? (g0 + log1pf(__expf(-g0))) : log1pf(__expf(g0));
  float gamma = -sp;

  // ---- Phase 1: scores strip = (Q K^T)/sqrt(DH). Q fragments are invariant
  // across key tiles -> hoist; 8 waves stride over 64 key tiles.
  {
    v16bf a0 = load_a_v(qbh, DH_, q0, 0,  lane);
    v16bf a1 = load_a_v(qbh, DH_, q0, 32, lane);
    for (int kt = wave; kt < S_ / 16; kt += 8) {
      v8f acc = {};
      acc = wmma_bf16(a0, load_bT_v(kbh, DH_, 0,  kt * 16, lane), acc);
      acc = wmma_bf16(a1, load_bT_v(kbh, DH_, 32, kt * 16, lane), acc);
      int n = lane & 15, mb = (lane >> 4) * 8;
#pragma unroll
      for (int rr = 0; rr < 8; ++rr)
        sc[(size_t)(mb + rr) * S_ + kt * 16 + n] = acc[rr] * 0.125f;  // 1/sqrt(64)
    }
  }
  __syncthreads();

  // ---- Phase 2: masked softmax -> cumsum decay -> second softmax ------------
  const int r = tid >> 4;        // row 0..15
  const int c = tid & 15;        // 16 workers per row
  const int Q = q0 + r;          // global query index (causal: j <= Q valid)
  const int j0 = c * (S_ / 16);  // 64 contiguous columns per worker
  float* row = sc + (size_t)r * S_;

  // softmax #1: max
  float mx = -3.0e38f;
  for (int j = j0; j < j0 + 64; ++j)
    mx = fmaxf(mx, (j <= Q) ? row[j] : -3.0e38f);
  red[r * 16 + c] = mx;
  __syncthreads();
  float rmx = -3.0e38f;
  for (int t = 0; t < 16; ++t) rmx = fmaxf(rmx, red[r * 16 + t]);
  __syncthreads();

  // softmax #1: sum of exp
  float sloc = 0.f;
  for (int j = j0; j < j0 + 64; ++j)
    sloc += (j <= Q) ? __expf(row[j] - rmx) : 0.f;
  red[r * 16 + c] = sloc;
  __syncthreads();
  float tot = 0.f, pre = 0.f;
  for (int t = 0; t < 16; ++t) {
    float v = red[r * 16 + t];
    if (t < c) pre += v;
    tot += v;
  }
  __syncthreads();

  // blocked cumsum of sm, distance decay, rescore (disttotal == 1)
  float inv = 1.0f / tot;
  float run = pre * inv;
  for (int j = j0; j < j0 + 64; ++j) {
    float sraw = row[j];
    float smv = (j <= Q) ? __expf(sraw - rmx) * inv : 0.f;
    run += smv;
    float pe = fabsf((float)(Q - j));
    float ds = sqrtf(fmaxf((1.0f - run) * pe, 0.f));
    float te = __expf(ds * gamma);
    te = fminf(fmaxf(te, 1e-5f), 1e5f);
    row[j] = (j <= Q) ? sraw * te : -1.0e32f;
  }
  __syncthreads();

  // softmax #2
  float mx2 = -3.0e38f;
  for (int j = j0; j < j0 + 64; ++j) mx2 = fmaxf(mx2, row[j]);
  red[r * 16 + c] = mx2;
  __syncthreads();
  float rmx2 = -3.0e38f;
  for (int t = 0; t < 16; ++t) rmx2 = fmaxf(rmx2, red[r * 16 + t]);
  __syncthreads();

  float s2 = 0.f;
  for (int j = j0; j < j0 + 64; ++j) s2 += __expf(row[j] - rmx2);
  red[r * 16 + c] = s2;
  __syncthreads();
  float tot2 = 0.f;
  for (int t = 0; t < 16; ++t) tot2 += red[r * 16 + t];
  __syncthreads();

  float inv2 = 1.0f / tot2;
  float zrow = (Q == 0) ? 0.f : 1.f;  // zero_pad=True: first query row zeroed
  float* arow = attn_out + (((size_t)b * H_ + h) * S_ + Q) * S_;
  for (int j = j0; j < j0 + 64; ++j) {
    float av = __expf(row[j] - rmx2) * inv2 * zrow;
    row[j] = av;                           // keep in LDS for attn @ V
    __builtin_nontemporal_store(av, arow + j);  // 134 MB stream: don't pollute L2
  }
  __syncthreads();

  // ---- Phase 3: out tile = attn(16x1024) @ V(1024x64). V^T layout makes
  // B-fragments contiguous along K. 4 waves, one dh-tile each.
  if (wave < 4) {
    const int n0 = wave * 16;
    v8f acc = {};
    for (int kk = 0; kk < S_; kk += 32) {
      v16bf a  = load_a_lds(sc, S_, kk, lane);
      v16bf bb = load_bT_v(vbhT, S_, kk, n0, lane);  // b(k,n) = V^T[n][k]
      acc = wmma_bf16(a, bb, acc);
    }
    int n = lane & 15, mb = (lane >> 4) * 8;
#pragma unroll
    for (int rr = 0; rr < 8; ++rr)
      concat[(size_t)(b * S_ + q0 + mb + rr) * D_ + h * DH_ + n0 + n] =
          (__bf16)acc[rr];
  }
}

// ---- Host-side orchestration ------------------------------------------------
extern "C" void kernel_launch(void* const* d_in, const int* in_sizes, int n_in,
                              void* d_out, int out_size, void* d_ws, size_t ws_size,
                              hipStream_t stream) {
  (void)in_sizes; (void)n_in; (void)out_size; (void)ws_size;
  const float* q      = (const float*)d_in[0];
  const float* k      = (const float*)d_in[1];
  const float* v      = (const float*)d_in[2];
  /* d_in[3] = mask: causal tril, recomputed from indices on device */
  const float* Wq     = (const float*)d_in[4];
  const float* bq     = (const float*)d_in[5];
  const float* Wk     = (const float*)d_in[6];
  const float* bk     = (const float*)d_in[7];
  const float* Wv     = (const float*)d_in[8];
  const float* bv     = (const float*)d_in[9];
  const float* Wo     = (const float*)d_in[10];
  const float* bo     = (const float*)d_in[11];
  const float* gammas = (const float*)d_in[12];

  // Workspace layout (all bf16)
  const size_t xe = (size_t)B_ * S_ * D_;   // 2,097,152 activation elems
  const size_t we = (size_t)D_ * D_;        //   262,144 weight elems
  bf16* p = (bf16*)d_ws;
  bf16* qb  = p;  p += xe;
  bf16* kb  = p;  p += xe;
  bf16* vb  = p;  p += xe;
  bf16* Wqb = p;  p += we;
  bf16* Wkb = p;  p += we;
  bf16* Wvb = p;  p += we;
  bf16* Wob = p;  p += we;
  bf16* qh  = p;  p += xe;                  // [B,H,S,DH]
  bf16* kh  = p;  p += xe;                  // [B,H,S,DH]
  bf16* vhT = p;  p += xe;                  // [B,H,DH,S]
  bf16* concat = p;                         // [B*S, D]

  float* out      = (float*)d_out;                  // [B,S,D]
  float* attn_out = out + (size_t)B_ * S_ * D_;     // [B,H,S,S]

  // fp32 -> bf16 streaming conversions (one-time; enables b128 fragment loads)
  {
    dim3 cb(256, 1, 1);
    int xa8 = (int)(xe / 8), wa8 = (int)(we / 8);
    f32_to_bf16_kernel<<<dim3((xa8 + 255) / 256), cb, 0, stream>>>(q,  qb,  xa8);
    f32_to_bf16_kernel<<<dim3((xa8 + 255) / 256), cb, 0, stream>>>(k,  kb,  xa8);
    f32_to_bf16_kernel<<<dim3((xa8 + 255) / 256), cb, 0, stream>>>(v,  vb,  xa8);
    f32_to_bf16_kernel<<<dim3((wa8 + 255) / 256), cb, 0, stream>>>(Wq, Wqb, wa8);
    f32_to_bf16_kernel<<<dim3((wa8 + 255) / 256), cb, 0, stream>>>(Wk, Wkb, wa8);
    f32_to_bf16_kernel<<<dim3((wa8 + 255) / 256), cb, 0, stream>>>(Wv, Wvb, wa8);
    f32_to_bf16_kernel<<<dim3((wa8 + 255) / 256), cb, 0, stream>>>(Wo, Wob, wa8);
  }

  dim3 gblk(32, 1, 1);
  dim3 ggrid((B_ * S_) / 16, D_ / 64, 1);  // 256 x 8 wave-tiles (16x64 each)

  // Q/K projections -> bf16 head layout; V -> bf16 transposed head layout
  proj_gemm_kernel<<<ggrid, gblk, 0, stream>>>(qb, Wqb, bq, qh,  nullptr, D_, D_, 0);
  proj_gemm_kernel<<<ggrid, gblk, 0, stream>>>(kb, Wkb, bk, kh,  nullptr, D_, D_, 0);
  proj_gemm_kernel<<<ggrid, gblk, 0, stream>>>(vb, Wvb, bv, vhT, nullptr, D_, D_, 2);

  // Fused attention: scores -> decay -> softmax -> attn out + attn@V
  dim3 ablk(256, 1, 1);
  dim3 agrid(S_ / 16, H_, B_);
  size_t smem = (size_t)(16 * S_ + 16 * 16) * sizeof(float);  // 66,560 B < 320 KB
  attn_kernel<<<agrid, ablk, smem, stream>>>(qh, kh, vhT, gammas, attn_out, concat);

  // Output projection -> fp32 d_out
  proj_gemm_kernel<<<ggrid, gblk, 0, stream>>>(concat, Wob, bo, nullptr, out, D_, D_, 1);
}
